// BasicBlock_1w4a_LUT_13804024889756
// MI455X (gfx1250) — 16-bit and 8-bit matrix operand order repaired and checked by reading, compiled, not run on hardware
//
#include <hip/hip_runtime.h>
#include <hip/hip_bf16.h>

typedef __attribute__((ext_vector_type(16))) __bf16 v16bf;
typedef __attribute__((ext_vector_type(8)))  float  v8f;
typedef __attribute__((ext_vector_type(8)))  int    v8i;

#define BATCH 16
#define CIN   32
#define CC    32
#define HH    224
#define WW    224
#define K1    288
#define TPX   112
#define TCOL  114
#define NWAVE 7
#define NTHR  (NWAVE*32)
#define XSTR  36
#define ASTRD 10

#define OFF_W1   0
#define OFF_W2   18432
#define OFF_LUT  28672
#define OFF_ACT  29696

#if defined(__has_builtin)
#  if __has_builtin(__builtin_amdgcn_global_load_async_to_lds_b32) && \
      __has_builtin(__builtin_amdgcn_s_wait_asynccnt)
#    define USE_ASYNC_LDS 1
#  endif
#endif
#ifndef USE_ASYNC_LDS
#  define USE_ASYNC_LDS 0
#endif

#if USE_ASYNC_LDS
#define ASYNC_B32(gptr, lptr)                                              \
    __builtin_amdgcn_global_load_async_to_lds_b32(                         \
        (__attribute__((address_space(1))) int*)(gptr),                    \
        (__attribute__((address_space(3))) int*)(lptr), 0, 0)
#endif

__global__ __launch_bounds__(256) void bb_prep_kernel(
    const float* __restrict__ w1f, const float* __restrict__ w2f,
    const float* __restrict__ bn1w, const float* __restrict__ bn1b,
    const float* __restrict__ bn1m, const float* __restrict__ bn1v,
    const float* __restrict__ bn2w, const float* __restrict__ bn2b,
    const float* __restrict__ bn2m, const float* __restrict__ bn2v,
    const float* __restrict__ a1p, const float* __restrict__ a2p,
    const float* __restrict__ nsp,
    __bf16* __restrict__ w1frag, signed char* __restrict__ w2frag,
    float* __restrict__ luts)
{
    __shared__ __bf16      wtmp1[288 * 32];
    __shared__ signed char wtmp2[320 * 32];

    const int t = threadIdx.x;
    if (t < 32) {
        float mean = 0.f;
        for (int k = 0; k < K1; ++k) mean += w1f[t * K1 + k];
        mean *= (1.0f / (float)K1);
        for (int ko = 0; ko < K1; ++ko) {
            int ci = ko / 9, r = ko - ci * 9;
            float d = w1f[t * K1 + ko] - mean;
            float s = (d > 0.f) ? 1.f : ((d < 0.f) ? -1.f : 0.f);
            wtmp1[(r * 32 + ci) * 32 + t] = (__bf16)s;
        }
        const float a1 = a1p[0], a2 = a2p[0], ns = nsp[0];
        {
            float stdv = sqrtf(bn1v[t] + 1e-5f);
            float wc = bn1w[t] / stdv;
            float bc = bn1b[t] - wc * bn1m[t];
            float inv = 1.0f / (a1 * wc);
            float t0 = rintf((0.5f * a2 - bc) * inv);
            float t1 = rintf((1.5f * a2 - bc) * inv);
            luts[t] = t0;  luts[32 + t] = t1 - t0;
        }
        {
            float stdv = sqrtf(bn2v[t] + 1e-5f);
            float wc = bn2w[t] / stdv;
            float bc = bn2b[t] - wc * bn2m[t];
            float inv = 1.0f / (a2 * wc);
            float t0 = rintf((0.5f * ns - bc) * inv);
            float t1 = rintf((1.5f * ns - bc) * inv);
            luts[64 + t] = t0;  luts[96 + t] = t1 - t0;
        }
    } else if (t < 64) {
        const int c = t - 32;
        float mean = 0.f;
        for (int k = 0; k < K1; ++k) mean += w2f[c * K1 + k];
        mean *= (1.0f / (float)K1);
        for (int ko = 0; ko < K1; ++ko) {
            int ci = ko / 9, r = ko - ci * 9;
            float d = w2f[c * K1 + ko] - mean;
            signed char s = (d > 0.f) ? 1 : ((d < 0.f) ? -1 : 0);
            wtmp2[(r * 32 + ci) * 32 + c] = s;
        }
        for (int k = K1; k < 320; ++k) wtmp2[k * 32 + c] = 0;
    }
    __syncthreads();

    for (int e = t; e < 9 * 2 * 32 * 16; e += 256) {
        int i  = e & 15;
        int L  = (e >> 4) & 31;
        int nt = (e >> 9) & 1;
        int ks = e >> 10;
        int k  = ks * 32 + ((i & 8) << 1) + (L >> 4) * 8 + (i & 7);
        int n  = nt * 16 + (L & 15);
        w1frag[e] = wtmp1[k * 32 + n];
    }
    for (int e = t; e < 5 * 2 * 32 * 32; e += 256) {
        int j  = e & 31;
        int L  = (e >> 5) & 31;
        int nt = (e >> 10) & 1;
        int ks = e >> 11;
        int vv = j >> 2, b4 = j & 3;
        int k  = ks * 64 + ((vv >> 2) << 5) + ((L >> 4) << 4) + ((vv & 3) << 2) + b4;
        int n  = nt * 16 + (L & 15);
        w2frag[e] = wtmp2[k * 32 + n];
    }
}

__global__ __launch_bounds__(NTHR) void bb_conv1_lut_kernel(
    const float* __restrict__ x, const __bf16* __restrict__ w1frag,
    const float* __restrict__ luts, unsigned char* __restrict__ act)
{
    __shared__ float         lds_x[3 * TCOL * XSTR];
    __shared__ unsigned char trp1[NWAVE][16][32];

    const int tid = threadIdx.x;
    const int wv  = tid >> 5;
    const int L   = tid & 31;
    const int lh  = L >> 4;
    const int lm  = L & 15;

    int blk = blockIdx.x;
    const int xhalf = blk & 1;          blk >>= 1;
    const int y     = blk % HH;
    const int bb    = blk / HH;
    const int X0    = xhalf * TPX;

    for (int idx = tid; idx < CIN * 3 * TCOL; idx += NTHR) {
        int ci  = idx / (3 * TCOL);
        int rem = idx - ci * (3 * TCOL);
        int rr  = rem / TCOL;
        int cc  = rem - rr * TCOL;
        int yy  = y + rr - 1;
        int xx  = X0 - 1 + cc;
        bool inb = (yy >= 0) & (yy < HH) & (xx >= 0) & (xx < WW);
        const int loff = (rr * TCOL + cc) * XSTR + ci;
#if USE_ASYNC_LDS
        if (inb) {
            const float* gp = x + (((size_t)bb * CIN + ci) * HH + yy) * WW + xx;
            ASYNC_B32(gp, &lds_x[loff]);
        } else {
            lds_x[loff] = 0.f;
        }
#else
        int yyc = min(max(yy, 0), HH - 1);
        int xxc = min(max(xx, 0), WW - 1);
        float v = x[(((size_t)bb * CIN + ci) * HH + yyc) * WW + xxc];
        lds_x[loff] = inb ? v : 0.f;
#endif
    }
#if USE_ASYNC_LDS
    __builtin_amdgcn_s_wait_asynccnt(0);
#endif
    __syncthreads();

    const v16bf* __restrict__ bfragv = (const v16bf*)w1frag;

    v8f acc0 = {};
    v8f acc1 = {};

    #pragma unroll
    for (int ks = 0; ks < 9; ++ks) {
        const int kh = ks / 3, kw = ks - kh * 3;
        const int cb = (kh * TCOL + wv * 16 + lm + kw) * XSTR + lh * 8;
        union { float4 q; float e[4]; } q0, q1, q2, q3;
        q0.q = *(const float4*)&lds_x[cb];
        q1.q = *(const float4*)&lds_x[cb + 4];
        q2.q = *(const float4*)&lds_x[cb + 16];
        q3.q = *(const float4*)&lds_x[cb + 20];
        union { v16bf v; __bf16 e[16]; } ah, al;
        #pragma unroll
        for (int i = 0; i < 4; ++i) {
            float v0 = q0.e[i], v1 = q1.e[i], v2 = q2.e[i], v3 = q3.e[i];
            __bf16 h0 = (__bf16)v0, h1 = (__bf16)v1, h2 = (__bf16)v2, h3 = (__bf16)v3;
            ah.e[i]      = h0;  al.e[i]      = (__bf16)(v0 - (float)h0);
            ah.e[4 + i]  = h1;  al.e[4 + i]  = (__bf16)(v1 - (float)h1);
            ah.e[8 + i]  = h2;  al.e[8 + i]  = (__bf16)(v2 - (float)h2);
            ah.e[12 + i] = h3;  al.e[12 + i] = (__bf16)(v3 - (float)h3);
        }
        v16bf bf0 = bfragv[(ks * 2 + 0) * 32 + L];
        v16bf bf1 = bfragv[(ks * 2 + 1) * 32 + L];
        acc0 = __builtin_amdgcn_wmma_f32_16x16x32_bf16(false, ah.v, false, bf0, (short)0, acc0, false, false);
        acc0 = __builtin_amdgcn_wmma_f32_16x16x32_bf16(false, al.v, false, bf0, (short)0, acc0, false, false);
        acc1 = __builtin_amdgcn_wmma_f32_16x16x32_bf16(false, ah.v, false, bf1, (short)0, acc1, false, false);
        acc1 = __builtin_amdgcn_wmma_f32_16x16x32_bf16(false, al.v, false, bf1, (short)0, acc1, false, false);
    }

    union { v8f v; float e[8]; } A0, A1;
    A0.v = acc0; A1.v = acc1;
    const float t0a = luts[lm],      dfa = luts[32 + lm];
    const float t0b = luts[16 + lm], dfb = luts[48 + lm];
    #pragma unroll
    for (int r = 0; r < 8; ++r) {
        const int p = r + lh * 8;
        float lv0 = 0.f, lv1 = 0.f;
        #pragma unroll
        for (int j = 0; j < 7; ++j) {
            float th0 = t0a + (float)j * dfa;
            float th1 = t0b + (float)j * dfb;
            if (j & 1) { lv0 += (A0.e[r] >= th0); lv1 += (A1.e[r] >= th1); }
            else       { lv0 += (A0.e[r] >  th0); lv1 += (A1.e[r] >  th1); }
        }
        trp1[wv][p][lm]      = (unsigned char)lv0;
        trp1[wv][p][16 + lm] = (unsigned char)lv1;
    }
    {
        const int p2   = L >> 1;
        const int half = L & 1;
        uint4 dat = *(const uint4*)&trp1[wv][p2][half * 16];
        unsigned char* gb = act +
            ((((size_t)bb * HH + y) * WW + X0 + wv * 16 + p2) * 32 + half * 16);
        *(uint4*)gb = dat;
    }
}

__global__ __launch_bounds__(NTHR) void bb_conv2_lut_kernel(
    const unsigned char* __restrict__ act, const signed char* __restrict__ w2frag,
    const float* __restrict__ luts, float* __restrict__ out)
{
    __shared__ unsigned int lds_a[3 * TCOL * ASTRD];
    __shared__ float        trp2[NWAVE][16][32];

    const int tid = threadIdx.x;
    const int wv  = tid >> 5;
    const int L   = tid & 31;
    const int lh  = L >> 4;
    const int lm  = L & 15;

    int blk = blockIdx.x;
    const int xhalf = blk & 1;          blk >>= 1;
    const int y     = blk % HH;
    const int bb    = blk / HH;
    const int X0    = xhalf * TPX;

    for (int d = tid; d < 3 * TCOL * 8; d += NTHR) {
        int kh  = d / (TCOL * 8);
        int rem = d - kh * (TCOL * 8);
        int cx  = rem >> 3;
        int w8  = rem & 7;
        int yy  = y + kh - 1;
        int xx  = X0 - 1 + cx;
        bool inb = (yy >= 0) & (yy < HH) & (xx >= 0) & (xx < WW);
        const int loff = (kh * TCOL + cx) * ASTRD + w8;
#if USE_ASYNC_LDS
        if (inb) {
            const unsigned char* gp = act + ((((size_t)bb * HH + yy) * WW + xx) * 32 + w8 * 4);
            ASYNC_B32(gp, &lds_a[loff]);
        } else {
            lds_a[loff] = 0u;
        }
#else
        int yyc = min(max(yy, 0), HH - 1);
        int xxc = min(max(xx, 0), WW - 1);
        unsigned int v = *(const unsigned int*)
            (act + ((((size_t)bb * HH + yyc) * WW + xxc) * 32 + w8 * 4));
        lds_a[loff] = inb ? v : 0u;
#endif
    }
#if USE_ASYNC_LDS
    __builtin_amdgcn_s_wait_asynccnt(0);
#endif
    __syncthreads();

    const v8i* __restrict__ bfragv = (const v8i*)w2frag;

    v8i acc0 = {};
    v8i acc1 = {};

    #pragma unroll
    for (int ks = 0; ks < 5; ++ks) {
        union { v8i v; unsigned int w[8]; } af;
        #pragma unroll
        for (int half = 0; half < 2; ++half) {
            const int r = ks * 2 + half;
            if (r < 9) {
                const int kh = r / 3, kw = r - kh * 3;
                const int bdw = (kh * TCOL + wv * 16 + lm + kw) * ASTRD + lh * 4;
                uint2 d0 = *(const uint2*)&lds_a[bdw];
                uint2 d1 = *(const uint2*)&lds_a[bdw + 2];
                af.w[half * 4 + 0] = d0.x;
                af.w[half * 4 + 1] = d0.y;
                af.w[half * 4 + 2] = d1.x;
                af.w[half * 4 + 3] = d1.y;
            } else {
                af.w[half * 4 + 0] = 0u; af.w[half * 4 + 1] = 0u;
                af.w[half * 4 + 2] = 0u; af.w[half * 4 + 3] = 0u;
            }
        }
        v8i bf0 = bfragv[(ks * 2 + 0) * 32 + L];
        v8i bf1 = bfragv[(ks * 2 + 1) * 32 + L];
        acc0 = __builtin_amdgcn_wmma_i32_16x16x64_iu8(false, af.v, true, bf0, acc0, false, false);
        acc1 = __builtin_amdgcn_wmma_i32_16x16x64_iu8(false, af.v, true, bf1, acc1, false, false);
    }

    union { v8i v; int e[8]; } A0, A1;
    A0.v = acc0; A1.v = acc1;
    const float t0a = luts[64 + lm], dfa = luts[96 + lm];
    const float t0b = luts[80 + lm], dfb = luts[112 + lm];
    #pragma unroll
    for (int r = 0; r < 8; ++r) {
        const int p = r + lh * 8;
        float xv0 = (float)A0.e[r];
        float xv1 = (float)A1.e[r];
        float lv0 = 0.f, lv1 = 0.f;
        #pragma unroll
        for (int j = 0; j < 7; ++j) {
            float th0 = t0a + (float)j * dfa;
            float th1 = t0b + (float)j * dfb;
            if (j & 1) { lv0 += (xv0 >= th0); lv1 += (xv1 >= th1); }
            else       { lv0 += (xv0 >  th0); lv1 += (xv1 >  th1); }
        }
        trp2[wv][p][lm]      = lv0;
        trp2[wv][p][16 + lm] = lv1;
    }
    {
        const int ch = L;
        float* obase = out + (((size_t)bb * CC + ch) * HH + y) * WW + X0 + wv * 16;
        #pragma unroll
        for (int q = 0; q < 4; ++q) {
            float4 o;
            o.x = trp2[wv][q * 4 + 0][ch];
            o.y = trp2[wv][q * 4 + 1][ch];
            o.z = trp2[wv][q * 4 + 2][ch];
            o.w = trp2[wv][q * 4 + 3][ch];
            *(float4*)(obase + q * 4) = o;
        }
    }
}

extern "C" void kernel_launch(void* const* d_in, const int* in_sizes, int n_in,
                              void* d_out, int out_size, void* d_ws, size_t ws_size,
                              hipStream_t stream) {
    (void)in_sizes; (void)n_in; (void)out_size; (void)ws_size;

    const float* x    = (const float*)d_in[0];
    const float* w1f  = (const float*)d_in[1];
    const float* w2f  = (const float*)d_in[2];
    const float* bn1w = (const float*)d_in[3];
    const float* bn1b = (const float*)d_in[4];
    const float* bn1m = (const float*)d_in[5];
    const float* bn1v = (const float*)d_in[6];
    const float* bn2w = (const float*)d_in[7];
    const float* bn2b = (const float*)d_in[8];
    const float* bn2m = (const float*)d_in[9];
    const float* bn2v = (const float*)d_in[10];
    const float* a1p  = (const float*)d_in[11];
    const float* a2p  = (const float*)d_in[12];
    const float* nsp  = (const float*)d_in[13];

    char* ws = (char*)d_ws;
    __bf16*        w1frag = (__bf16*)(ws + OFF_W1);
    signed char*   w2frag = (signed char*)(ws + OFF_W2);
    float*         luts   = (float*)(ws + OFF_LUT);
    unsigned char* actb   = (unsigned char*)(ws + OFF_ACT);

    bb_prep_kernel<<<1, 256, 0, stream>>>(w1f, w2f,
        bn1w, bn1b, bn1m, bn1v, bn2w, bn2b, bn2m, bn2v,
        a1p, a2p, nsp, w1frag, w2frag, luts);

    const int nblocks = BATCH * HH * (WW / TPX);
    dim3 blk(NTHR), grd(nblocks);
    bb_conv1_lut_kernel<<<grd, blk, 0, stream>>>(x, w1frag, luts, actb);
    bb_conv2_lut_kernel<<<grd, blk, 0, stream>>>(actb, w2frag, luts, (float*)d_out);
}
